// StructNConv2D_e_with_d_39737037423459
// MI455X (gfx1250) — matrix operand order repaired and checked by reading, not compiled, not run on hardware
//
#include <hip/hip_runtime.h>
#include <hip/hip_bf16.h>
#include <stdint.h>

#define EPSF 1e-20f

static constexpr int B_   = 2;
static constexpr int CIN  = 64;
static constexpr int H_   = 192;
static constexpr int W_   = 192;
static constexpr int C4   = CIN * 4;        // 256 (in channels for GEMM)
static constexpr int O4   = 256;            // COUT*4 (out channels for GEMM)
static constexpr int HW   = H_ * W_;        // 36864
static constexpr long long NOUT = (long long)B_ * O4 * HW;   // elems in s_out

// workspace layout (bytes)
static constexpr size_t WS_CWBF = 256;                             // 256*256 bf16
static constexpr size_t WS_CSXS = WS_CWBF + (size_t)O4 * C4 * 2;   // B*256*HW bf16
static constexpr size_t WS_CSP  = WS_CSXS + (size_t)B_ * C4 * HW * 2;

// LDS layout for K2 (bytes): [weights 131072][Xn 64*264*2][Xd 64*264*2]
static constexpr int XPITCH   = 264;                       // bf16 elems, bank-spread pad
static constexpr int SMEM_K2  = 131072 + 2 * 64 * XPITCH * 2;  // 198656 B < 320 KB/WGP

typedef __attribute__((ext_vector_type(16))) __bf16       v16bf;
typedef __attribute__((ext_vector_type(8)))  float        v8f;
typedef __attribute__((ext_vector_type(4)))  unsigned int u32x4;

union BfVec { v16bf v; u32x4 u[2]; };   // ext vectors only -> trivially constructible

__device__ __forceinline__ unsigned short f2bf(float f) {
    unsigned int u = __float_as_uint(f);
    u = (u + 0x7FFFu + ((u >> 16) & 1u)) >> 16;   // RNE
    return (unsigned short)u;
}
__device__ __forceinline__ float bf2f(unsigned short h) {
    return __uint_as_float(((unsigned int)h) << 16);
}
__device__ __forceinline__ float rcpf(float x) {            // 1 ULP v_rcp_f32
    return __builtin_amdgcn_rcpf(x);
}

// ---------------------------------------------------------------------------
// K0: weight sums + bf16 conversion of channel weights
// ---------------------------------------------------------------------------
__global__ void k0_weights(const float* __restrict__ cw, const float* __restrict__ sw,
                           float* __restrict__ sums, unsigned short* __restrict__ cwBf) {
    __shared__ float red[256];
    const int t = threadIdx.x;
    float accC = 0.f;
    for (int i = t; i < O4 * C4; i += 256) { float v = cw[i]; accC += v; cwBf[i] = f2bf(v); }
    float accS = 0.f;
    for (int i = t; i < C4 * 9; i += 256) accS += sw[i];

    red[t] = accC; __syncthreads();
    for (int s = 128; s > 0; s >>= 1) { if (t < s) red[t] += red[t + s]; __syncthreads(); }
    if (t == 0) sums[0] = red[0];
    __syncthreads();
    red[t] = accS; __syncthreads();
    for (int s = 128; s > 0; s >>= 1) { if (t < s) red[t] += red[t + s]; __syncthreads(); }
    if (t == 0) sums[1] = red[0];
}

// ---------------------------------------------------------------------------
// K1: neighborhood + propagation -> bf16 NCHW {cs_prop*s_prop, cs_prop}
// (reference's argmax over a duplicated axis is always 0 -> only direction 0,
//  i.e. the (-1,-1)/(+1,+1) diagonal pair, contributes)
// ---------------------------------------------------------------------------
__global__ void k1_prep(const float* __restrict__ d,  const float* __restrict__ cd,
                        const float* __restrict__ s,  const float* __restrict__ cs,
                        const float* __restrict__ w_s_from_d, const float* __restrict__ w_prop,
                        unsigned short* __restrict__ csxs, unsigned short* __restrict__ csp) {
    long long gid = (long long)blockIdx.x * blockDim.x + threadIdx.x;  // over B*CIN*HW
    if (gid >= (long long)B_ * CIN * HW) return;
    const int  hw = (int)(gid % HW);
    const long long bc = gid / HW;
    const int  c = (int)(bc % CIN);
    const int  b = (int)(bc / CIN);
    const int  h = hw / W_, w = hw % W_;

    const float* dB  = d  + (size_t)(b * CIN + c) * HW;
    const float* cdB = cd + (size_t)(b * CIN + c) * HW;

    const bool tl = (h > 0) && (w > 0);
    const bool br = (h < H_ - 1) && (w < W_ - 1);
    const float mn  = tl ? dB [(h - 1) * W_ + (w - 1)] : 0.f;
    const float cmn = tl ? cdB[(h - 1) * W_ + (w - 1)] : 0.f;
    const float mx  = br ? dB [(h + 1) * W_ + (w + 1)] : 0.f;
    const float cmx = br ? cdB[(h + 1) * W_ + (w + 1)] : 0.f;

    float r0 = mn * rcpf(mx + EPSF);
    r0 = fminf(fmaxf(r0, 0.f), 1.f);
    const float c0  = cmn * cmx;                 // cs_from_d
    const float wsd = w_s_from_d[0];
    const float sfd = (1.f - wsd) * r0 + wsd * r0 * r0;   // s_from_d
    const float wp  = w_prop[c];
    const float invWp1 = rcpf(wp + 1.f);

    const float* sB  = s  + ((size_t)(b * CIN + c) * 4) * HW + hw;
    const float* csB = cs + ((size_t)(b * CIN + c) * 4) * HW + hw;
    const size_t outBase = (size_t)(b * C4 + c * 4) * HW + hw;
#pragma unroll
    for (int k = 0; k < 4; ++k) {
        const float sk = sB [(size_t)k * HW];
        const float ck = csB[(size_t)k * HW];
        const float sp = (wp * ck * sk + c0 * sfd) * rcpf(wp * ck + c0 + EPSF);
        const float cp = (wp * ck + c0) * invWp1;
        csxs[outBase + (size_t)k * HW] = f2bf(cp * sp);
        csp [outBase + (size_t)k * HW] = f2bf(cp);
    }
}

// ---------------------------------------------------------------------------
// K2: fused depthwise-3x3 normalized conv + channel-mix GEMM via bf16 WMMA.
// Block: 256 threads (8 wave32), 64 consecutive pixels of one image row.
// Weights DMA'd to LDS with GLOBAL_LOAD_ASYNC_TO_LDS_B128 (ASYNCcnt), hidden
// behind the conv phase. GEMM register-blocked 2(o-tiles) x 2(px-tiles) per
// wave: A shared across px-tiles, B shared across o-tiles -> 12 ds loads per
// 8 WMMAs. D laid out [o][px] so f32 epilogue stores coalesce over pixels.
// ---------------------------------------------------------------------------
__global__ void k2_conv_gemm(const unsigned short* __restrict__ csxs,
                             const unsigned short* __restrict__ csp,
                             const unsigned short* __restrict__ cwBf,
                             const float* __restrict__ sw,
                             const float* __restrict__ sums,
                             float* __restrict__ out) {
    extern __shared__ char smem[];
    u32x4*          smW = (u32x4*)smem;                               // 256x256 bf16 weights
    unsigned short* Xn  = (unsigned short*)(smem + 131072);           // [64][XPITCH] bf16
    unsigned short* Xd  = Xn + 64 * XPITCH;

    const int t      = threadIdx.x;
    const int tile   = blockIdx.x;                 // 0..1151
    const int b      = tile / (HW / 64);
    const int hwBase = (tile % (HW / 64)) * 64;    // W_%64==0: tile stays in one row
    const int h      = hwBase / W_;
    const int w0     = hwBase % W_;

    // ---- async-DMA the full 256x256 bf16 weight matrix into LDS ------------
    // Per-lane: 16B from global saddr+vaddr to LDS at vdst, tracked by ASYNCcnt;
    // overlaps with the conv phase below.
    {
        const unsigned ldsBase = (unsigned)(uintptr_t)(void*)smW;   // LDS byte offset
#pragma unroll
        for (int it = 0; it < 32; ++it) {
            const unsigned off  = (unsigned)(it * 256 + t) * 16u;   // byte offset
            const unsigned ldsA = ldsBase + off;
            asm volatile("global_load_async_to_lds_b128 %0, %1, %2"
                         :: "v"(ldsA), "v"(off), "s"(cwBf)
                         : "memory");
        }
    }

    // ---- depthwise conv -> pixel-major / channel-contiguous bf16 LDS tiles ----
    {
        const int   px    = t & 63;
        const int   cq    = t >> 6;                // 4 channel lanes
        const int   w     = w0 + px;
        const float invSW = rcpf(sums[1] + EPSF);
        for (int c = cq; c < C4; c += 4) {
            const unsigned short* aB  = csxs + (size_t)(b * C4 + c) * HW;
            const unsigned short* dB  = csp  + (size_t)(b * C4 + c) * HW;
            const float*          wr  = sw + c * 9;
            float nom = 0.f, den = 0.f;
#pragma unroll
            for (int dy = -1; dy <= 1; ++dy) {
                const int  hh  = h + dy;
                const bool hok = (hh >= 0) && (hh < H_);
#pragma unroll
                for (int dx = -1; dx <= 1; ++dx) {
                    const int ww = w + dx;
                    if (hok && ww >= 0 && ww < W_) {
                        const float wgt = wr[(dy + 1) * 3 + (dx + 1)];
                        const int   off = hh * W_ + ww;
                        nom += wgt * bf2f(aB[off]);
                        den += wgt * bf2f(dB[off]);
                    }
                }
            }
            const float ssp = nom * rcpf(den + EPSF);       // s_spatial
            const float csv = den * invSW;                  // cs_spatial
            Xn[px * XPITCH + c] = f2bf(csv * ssp);
            Xd[px * XPITCH + c] = f2bf(csv);
        }
    }
    // weight DMA must be complete and visible to all waves before GEMM
    asm volatile("s_wait_asynccnt 0" ::: "memory");
    __syncthreads();

    // ---- GEMM: 2x2 register-blocked 16x16 tiles, K=256 in 8 steps of 32 ----
    const int   lane   = t & 31;
    const int   wv     = t >> 5;
    const int   m      = lane & 15;
    const bool  hiHalf = lane >= 16;
    const float invCW  = rcpf(sums[0] + EPSF);

#pragma unroll 1
    for (int i = 0; i < 2; ++i) {
        const int o0  = ((wv >> 1) * 4 + i * 2) * 16;  // o-tiles: o0, o0+16
        const int pt0 = (wv & 1) * 2;                  // px-tiles: pt0, pt0+1
        const int px0 = pt0 * 16 + m;                  // B columns for this lane
        const int px1 = px0 + 16;

        v8f accN0 = {0.f,0.f,0.f,0.f,0.f,0.f,0.f,0.f}, accN1 = accN0,
            accN2 = accN0, accN3 = accN0;
        v8f accD0 = accN0, accD1 = accN0, accD2 = accN0, accD3 = accN0;
        // acc{N,D}[oi*2+pi]: 0:(o0,px0) 1:(o0,px1) 2:(o0+16,px0) 3:(o0+16,px1)

        const u32x4* wr0 = (const u32x4*)((const char*)smW + (size_t)(o0 + m) * 512);
        const u32x4* wr1 = (const u32x4*)((const char*)wr0 + 16 * 512);
        const unsigned short* bn0 = Xn + px0 * XPITCH;
        const unsigned short* bn1 = Xn + px1 * XPITCH;
        const unsigned short* bd0 = Xd + px0 * XPITCH;
        const unsigned short* bd1 = Xd + px1 * XPITCH;

#pragma unroll 1
        for (int k0 = 0; k0 < 256; k0 += 32) {
            BfVec A0, A1, Bn0, Bn1, Bd0, Bd1;
            // A layout (matches B): lane<16 -> K=k0..k0+15,
            //                       lane>=16 -> K=k0+16..k0+31
            const int ka = k0 + (hiHalf ? 16 : 0);
            A0.u[0] = wr0[ka >> 3];  A0.u[1] = wr0[(ka >> 3) + 1];
            A1.u[0] = wr1[ka >> 3];  A1.u[1] = wr1[(ka >> 3) + 1];
            // B layout (same as A, contiguous): lane<16 -> K=k0..k0+15, lane>=16 -> +16
            const int kb = k0 + (hiHalf ? 16 : 0);
            const u32x4* pn0 = (const u32x4*)(bn0 + kb);
            const u32x4* pn1 = (const u32x4*)(bn1 + kb);
            const u32x4* pd0 = (const u32x4*)(bd0 + kb);
            const u32x4* pd1 = (const u32x4*)(bd1 + kb);
            Bn0.u[0] = pn0[0]; Bn0.u[1] = pn0[1];
            Bn1.u[0] = pn1[0]; Bn1.u[1] = pn1[1];
            Bd0.u[0] = pd0[0]; Bd0.u[1] = pd0[1];
            Bd1.u[0] = pd1[0]; Bd1.u[1] = pd1[1];

            accN0 = __builtin_amdgcn_wmma_f32_16x16x32_bf16(false, A0.v, false, Bn0.v,
                                                            (short)0, accN0, false, false);
            accD0 = __builtin_amdgcn_wmma_f32_16x16x32_bf16(false, A0.v, false, Bd0.v,
                                                            (short)0, accD0, false, false);
            accN1 = __builtin_amdgcn_wmma_f32_16x16x32_bf16(false, A0.v, false, Bn1.v,
                                                            (short)0, accN1, false, false);
            accD1 = __builtin_amdgcn_wmma_f32_16x16x32_bf16(false, A0.v, false, Bd1.v,
                                                            (short)0, accD1, false, false);
            accN2 = __builtin_amdgcn_wmma_f32_16x16x32_bf16(false, A1.v, false, Bn0.v,
                                                            (short)0, accN2, false, false);
            accD2 = __builtin_amdgcn_wmma_f32_16x16x32_bf16(false, A1.v, false, Bd0.v,
                                                            (short)0, accD2, false, false);
            accN3 = __builtin_amdgcn_wmma_f32_16x16x32_bf16(false, A1.v, false, Bn1.v,
                                                            (short)0, accN3, false, false);
            accD3 = __builtin_amdgcn_wmma_f32_16x16x32_bf16(false, A1.v, false, Bd1.v,
                                                            (short)0, accD3, false, false);
        }

        // D layout: vgpr r, lanes 0-15 -> M=r, N=lane; lanes 16-31 -> M=r+8
        const int oB   = o0 + (hiHalf ? 8 : 0);
        const int hwA  = hwBase + pt0 * 16 + m;
#pragma unroll
        for (int r = 0; r < 8; ++r) {
            const long long i00 = (long long)(b * O4 + oB + r) * HW + hwA;       // (o0 ,px0)
            const long long i01 = i00 + 16;                                      // (o0 ,px1)
            const long long i10 = i00 + (long long)16 * HW;                      // (o0',px0)
            const long long i11 = i10 + 16;                                      // (o0',px1)
            float dn;
            dn = accD0[r]; out[i00] = accN0[r] * rcpf(dn + EPSF); out[NOUT + i00] = dn * invCW;
            dn = accD1[r]; out[i01] = accN1[r] * rcpf(dn + EPSF); out[NOUT + i01] = dn * invCW;
            dn = accD2[r]; out[i10] = accN2[r] * rcpf(dn + EPSF); out[NOUT + i10] = dn * invCW;
            dn = accD3[r]; out[i11] = accN3[r] * rcpf(dn + EPSF); out[NOUT + i11] = dn * invCW;
        }
    }
}

// ---------------------------------------------------------------------------
extern "C" void kernel_launch(void* const* d_in, const int* in_sizes, int n_in,
                              void* d_out, int out_size, void* d_ws, size_t ws_size,
                              hipStream_t stream) {
    const float* d          = (const float*)d_in[0];
    const float* cd         = (const float*)d_in[1];
    const float* s          = (const float*)d_in[2];
    const float* cs         = (const float*)d_in[3];
    const float* w_s_from_d = (const float*)d_in[4];
    const float* w_prop     = (const float*)d_in[5];
    const float* cw         = (const float*)d_in[6];
    const float* sw         = (const float*)d_in[7];
    float* out = (float*)d_out;

    float*          sums  = (float*)d_ws;
    unsigned short* cwBf  = (unsigned short*)((char*)d_ws + WS_CWBF);
    unsigned short* csxs  = (unsigned short*)((char*)d_ws + WS_CSXS);
    unsigned short* cspW  = (unsigned short*)((char*)d_ws + WS_CSP);

    k0_weights<<<1, 256, 0, stream>>>(cw, sw, sums, cwBf);

    const long long n1 = (long long)B_ * CIN * HW;           // 4,718,592
    k1_prep<<<(unsigned)((n1 + 255) / 256), 256, 0, stream>>>(d, cd, s, cs, w_s_from_d,
                                                              w_prop, csxs, cspW);

    const int nTiles = (B_ * HW) / 64;                       // 1152
    k2_conv_gemm<<<nTiles, 256, SMEM_K2, stream>>>(csxs, cspW, cwBf, sw, sums, out);
}
